// FeedbackModelPseudo_74311524155784
// MI455X (gfx1250) — compile-verified, matrix operand order checked by reading, not run on hardware
//
#include <hip/hip_runtime.h>
#include <math.h>

// ---------------- problem constants ----------------
#define B_   16
#define S_   1024
#define H_   1024
#define NH_  16
#define DH_  64
#define N_   32
#define G4_  4096   // 4*H

typedef __bf16 bf16_t;
typedef __attribute__((ext_vector_type(16))) __bf16 v16bf;
typedef __attribute__((ext_vector_type(8)))  float  v8f;

// ---------------- workspace layout (bytes) ----------------
constexpr size_t WS_BAR   = 0;                                   // 2 x u32 barrier state
constexpr size_t WS_SID   = 256;                                 // span_id[1024] i32
constexpr size_t WS_SLAST = WS_SID   + 4096;                     // span_last[1024] i32
constexpr size_t WS_SCNT  = WS_SLAST + 4096;                     // span_cnt[32] f32
constexpr size_t WS_BIAS  = 16384;                               // combined b_ih+b_hh [4096] f32
constexpr size_t WS_H     = WS_BIAS + 4096*4;                    // h double buffer 2x[16,1024] bf16
constexpr size_t WS_MEAN  = WS_H    + 2*B_*H_*2;                 // mean_vec [16,32,1024] f32
constexpr size_t WS_XLN   = WS_MEAN + (size_t)B_*N_*H_*4;        // post-LN1 f32 [512,1024]
constexpr size_t WS_XLNB  = WS_XLN  + (size_t)512*1024*4;        // post-LN1 bf16
constexpr size_t WS_Q     = WS_XLNB + (size_t)512*1024*2;        // q f32
constexpr size_t WS_K     = WS_Q    + (size_t)512*1024*4;        // k f32
constexpr size_t WS_V     = WS_K    + (size_t)512*1024*4;        // v f32
constexpr size_t WS_CTXB  = WS_V    + (size_t)512*1024*4;        // ctx bf16
constexpr size_t WS_ATTN  = WS_CTXB + (size_t)512*1024*2;        // out-proj f32
constexpr size_t WS_AOUT  = WS_ATTN + (size_t)512*1024*4;        // post-LN2 f32
constexpr size_t WS_WQB   = WS_AOUT + (size_t)512*1024*4;        // Wq bf16 [1024,1024]
constexpr size_t WS_WKB   = WS_WQB  + (size_t)1024*1024*2;
constexpr size_t WS_WVB   = WS_WKB  + (size_t)1024*1024*2;
constexpr size_t WS_WOB   = WS_WVB  + (size_t)1024*1024*2;
constexpr size_t WS_WIHB  = WS_WOB  + (size_t)1024*1024*2;       // W_ih bf16 [4096,1024]
constexpr size_t WS_WHHB  = WS_WIHB + (size_t)4096*1024*2;       // W_hh bf16 [4096,1024]
constexpr size_t WS_XB    = WS_WHHB + (size_t)4096*1024*2;       // x bf16 [16384,1024]
constexpr size_t WS_XGB   = WS_XB   + (size_t)16384*1024*2;      // xg bf16 [B,S,4H]

// ---------------- CDNA5 async-to-LDS helpers ----------------
// GLOBAL_LOAD_ASYNC_TO_LDS_B128: per-lane 16B DMA global -> LDS, tracked on ASYNCcnt.
__device__ __forceinline__ void async_load_b128(unsigned lds_off, const void* gptr) {
  asm volatile("global_load_async_to_lds_b128 %0, %1, off" ::"v"(lds_off), "v"(gptr)
               : "memory");
}
template <int N>
__device__ __forceinline__ void wait_asynccnt() {
#if __has_builtin(__builtin_amdgcn_s_wait_asynccnt)
  __builtin_amdgcn_s_wait_asynccnt(N);
#else
  asm volatile("s_wait_asynccnt %0" ::"i"(N) : "memory");
#endif
}
// generic-pointer -> LDS byte offset (LDS aperture: addr[31:0] is the LDS address)
__device__ __forceinline__ unsigned lds_offset(const void* p) {
  return (unsigned)(uintptr_t)p;
}

// ---------------- WMMA fragment helpers ----------------
// 16-bit A/B fragment layout (ISA 7.12.2):
//   lane l: row/col = l%16 ; K-half = 8*(l/16)
//   VGPR j holds K pair at:  (j<4?0:16) + 8*(l/16) + 2*(j%4)
__device__ __forceinline__ v16bf load_frag(const bf16_t* __restrict__ base,
                                           int row, int ldk, int k0, int lane) {
  union { v16bf v; unsigned u[8]; } f;
  const unsigned* p = (const unsigned*)base;
  const int kh = (lane >> 4) << 3;
  const int rb = row * ldk + k0;
#pragma unroll
  for (int j = 0; j < 8; ++j) {
    int koff = ((j & 4) << 2) + kh + ((j & 3) << 1);
    f.u[j] = p[(rb + koff) >> 1];
  }
  return f.v;
}

// A fragment from a 16x32 bf16 LDS tile (row-major, 64B rows)
__device__ __forceinline__ v16bf load_frag_lds(const bf16_t* tile, int lane) {
  union { v16bf v; unsigned u[8]; } f;
  const unsigned* p = (const unsigned*)tile;
  const int kh = (lane >> 4) << 3;
  const int m  = lane & 15;
#pragma unroll
  for (int j = 0; j < 8; ++j) {
    int koff = ((j & 4) << 2) + kh + ((j & 3) << 1);
    f.u[j] = p[(m * 32 + koff) >> 1];
  }
  return f.v;
}

__device__ __forceinline__ v8f wmma_bf16(v16bf a, v16bf b, v8f c) {
  return __builtin_amdgcn_wmma_f32_16x16x32_bf16(false, a, false, b, (short)0, c,
                                                 false, false);
}

// branch-free saturating nonlinearities built on hardware v_exp_f32 / v_rcp_f32
// (no IEEE div_scale/div_fixup chains, no libm slow paths; co-executes with WMMA)
__device__ __forceinline__ float sigf(float x) {
  return __builtin_amdgcn_rcpf(1.0f + __expf(-x));     // +inf -> 1, -inf -> 0
}
__device__ __forceinline__ float tanh_fast(float x) {
  float t = __expf(2.0f * x);                          // +inf -> inf, -inf -> 0
  return 1.0f - 2.0f * __builtin_amdgcn_rcpf(t + 1.0f); // -> +1 / -1, NaN-free
}

// ---------------- device-wide barrier (graph-capture safe) ----------------
__device__ __forceinline__ void grid_barrier(unsigned* cnt, unsigned* gen, unsigned nblk) {
  __threadfence();
  __syncthreads();
  if (threadIdx.x == 0) {
    unsigned g = *(volatile unsigned*)gen;
    unsigned v = atomicAdd(cnt, 1u);
    if (v == nblk - 1u) {
      *(volatile unsigned*)cnt = 0u;
      __threadfence();
      atomicAdd(gen, 1u);
    } else {
      while (*(volatile unsigned*)gen == g) { __builtin_amdgcn_s_sleep(1); }
    }
  }
  __syncthreads();
}

// ---------------- small utility kernels ----------------
__global__ void cvt_kernel(const float* __restrict__ in, bf16_t* __restrict__ out, size_t n) {
  size_t i = (size_t)blockIdx.x * blockDim.x + threadIdx.x;
  size_t st = (size_t)gridDim.x * blockDim.x;
  for (; i < n; i += st) out[i] = (bf16_t)in[i];
}

__global__ void zero_f32_kernel(float* __restrict__ p, size_t n) {
  size_t i = (size_t)blockIdx.x * blockDim.x + threadIdx.x;
  size_t st = (size_t)gridDim.x * blockDim.x;
  for (; i < n; i += st) p[i] = 0.0f;
}

__global__ void bias_kernel(const float* __restrict__ a, const float* __restrict__ b,
                            float* __restrict__ out, int n) {
  int i = blockIdx.x * blockDim.x + threadIdx.x;
  if (i < n) out[i] = a[i] + b[i];
}

__global__ void init_kernel(const int* __restrict__ head, const int* __restrict__ tail,
                            int* __restrict__ span_id, int* __restrict__ span_last,
                            float* __restrict__ span_cnt, bf16_t* __restrict__ hbuf,
                            unsigned* __restrict__ bar) {
  int t = threadIdx.x;                    // 1 block x 1024 threads
  if (t < 2) bar[t] = 0u;
  if (t < N_) {
    float c = (float)(tail[t] - head[t] - 1);
    span_cnt[t] = (c > 0.f) ? c : 1.f;
  }
  if (t < S_) {
    int id = -1, last = 0;
    for (int n2 = 0; n2 < N_; ++n2) {
      if (t >= head[n2] + 1 && t < tail[n2]) { id = n2; if (t == tail[n2] - 1) last = 1; }
    }
    span_id[t] = id;
    span_last[t] = last;
  }
  for (int i = t; i < 2 * B_ * H_; i += 1024) hbuf[i] = (bf16_t)0.0f;
}

// ---------------- WMMA bf16 GEMM:  out[M,Nc] = A[M,K] * W[Nc,K]^T + bias ----------------
// grid = (M/16, Nc/128), block = 256 (8 waves, one 16x16 N-tile per wave).
// The 16x32 A tile (shared by all 8 waves) is double-buffered in LDS via
// GLOBAL_LOAD_ASYNC_TO_LDS_B128 issued by wave 0; s_wait_asynccnt paces the pipeline.
// OMODE: 0 -> fp32 output, 1 -> bf16 output (branch-free epilogue).
template <int OMODE>
__global__ __launch_bounds__(256) void wmma_gemm(const bf16_t* __restrict__ A,
                                                 const bf16_t* __restrict__ W,
                                                 const float* __restrict__ bias,
                                                 float* __restrict__ outF,
                                                 bf16_t* __restrict__ outB,
                                                 int M, int Nc, int K) {
  __shared__ bf16_t As[2][16 * 32];   // 2 x 1KB double buffer

  const int lane = threadIdx.x & 31;
  const int w    = threadIdx.x >> 5;
  const int mr   = blockIdx.x << 4;
  const int nc   = ((blockIdx.y << 3) + w) << 4;
  const int n    = lane & 15;
  const int hi8  = (lane >> 4) << 3;
  const int T    = K >> 5;

  const float bv = bias[nc + n];
  v8f acc;
#pragma unroll
  for (int r = 0; r < 8; ++r) acc[r] = bv;

  // async issue of A tile t into buffer t&1 (wave 0 only; 2 x b128 per lane = 1KB)
  auto issue = [&](int t) {
    const int row = threadIdx.x >> 2;         // 0..7
    const int seg = (threadIdx.x & 3) << 4;   // 16B segment
    const int k0  = t << 5;
    const char* g0 = (const char*)(A + (size_t)(mr + row) * K + k0) + seg;
    const char* g1 = (const char*)(A + (size_t)(mr + 8 + row) * K + k0) + seg;
    unsigned l = lds_offset(&As[t & 1][0]) + row * 64 + seg;
    async_load_b128(l, g0);
    async_load_b128(l + 8 * 64, g1);
  };

  if (threadIdx.x < 32) issue(0);

  for (int t = 0; t < T; ++t) {
    if (threadIdx.x < 32) {
      if (t + 1 < T) { issue(t + 1); wait_asynccnt<2>(); }
      else           { wait_asynccnt<0>(); }
    }
    __syncthreads();                       // A tile t visible to all waves
    if (t + 2 < T) {                       // prefetch B tile t+2 toward L2/L0
      __builtin_prefetch((const char*)&W[(size_t)(nc + n) * K + ((t + 2) << 5)], 0, 1);
    }
    v16bf a = load_frag_lds(&As[t & 1][0], lane);
    v16bf b = load_frag(W, nc + n, K, t << 5, lane);
    acc = wmma_bf16(a, b, acc);
    __syncthreads();                       // all waves done with tile t before re-fill
  }

#pragma unroll
  for (int r = 0; r < 8; ++r) {
    size_t idx = (size_t)(mr + r + hi8) * Nc + nc + n;
    if constexpr (OMODE == 0) outF[idx] = acc[r];
    else                      outB[idx] = (bf16_t)acc[r];
  }
}

// ---------------- persistent LSTM recurrence ----------------
// grid = 8 blocks x 256 threads = 64 waves; wave w owns hidden cols [16w,16w+16).
// Each step: 4 gate tiles of 16x16 (K=1024 -> 32 WMMA each), cell state in VGPRs,
// double-buffered h (bf16, L2-resident), one grid barrier per step, on-the-fly span pooling.
__global__ __launch_bounds__(256) void lstm_kernel(const bf16_t* __restrict__ xg,   // [B,S,4H]
                                                   const bf16_t* __restrict__ whh,  // [4H,H]
                                                   bf16_t* __restrict__ hbuf,       // 2x[B,H]
                                                   float* __restrict__ mean_vec,    // [B,N,H]
                                                   const int* __restrict__ span_id,
                                                   const int* __restrict__ span_last,
                                                   const float* __restrict__ span_cnt,
                                                   unsigned* __restrict__ bar_cnt,
                                                   unsigned* __restrict__ bar_gen) {
  const int lane = threadIdx.x & 31;
  const int wave = blockIdx.x * (blockDim.x >> 5) + (threadIdx.x >> 5); // 0..63
  const int u    = wave << 4;
  const int n    = lane & 15;
  const int hi8  = (lane >> 4) << 3;

  float c[8], pool[8];
#pragma unroll
  for (int r = 0; r < 8; ++r) { c[r] = 0.0f; pool[r] = 0.0f; }

  for (int s = 0; s < S_; ++s) {
    grid_barrier(bar_cnt, bar_gen, gridDim.x);   // h[s-1] globally visible
    const bf16_t* hr = hbuf + ((s + 1) & 1) * (B_ * H_);
    bf16_t*       hw = hbuf + (s & 1) * (B_ * H_);

    // init accumulators with precomputed input projection (biases already folded in)
    v8f acc[4];
#pragma unroll
    for (int gq = 0; gq < 4; ++gq) {
      const int go = gq * H_ + u + n;
#pragma unroll
      for (int r = 0; r < 8; ++r) {
        int m = r + hi8;
        acc[gq][r] = (float)xg[(size_t)(m * S_ + s) * G4_ + go];
      }
    }
    // gates += h_{s-1} * W_hh^T   (A frag reused across the 4 gate tiles)
    for (int k0 = 0; k0 < H_; k0 += 32) {
      v16bf a = load_frag(hr, lane & 15, H_, k0, lane);
#pragma unroll
      for (int gq = 0; gq < 4; ++gq) {
        v16bf b = load_frag(whh, gq * H_ + u + n, H_, k0, lane);
        acc[gq] = wmma_bf16(a, b, acc[gq]);
      }
    }
    // pointwise cell update + h store + span-mean pooling (branch-free nonlinearities)
    const int sid = span_id[s];
    const int slast = span_last[s];
#pragma unroll
    for (int r = 0; r < 8; ++r) {
      float ig = sigf(acc[0][r]);
      float fg = sigf(acc[1][r]);
      float gg = tanh_fast(acc[2][r]);
      float og = sigf(acc[3][r]);
      c[r] = fg * c[r] + ig * gg;
      float h = og * tanh_fast(c[r]);
      int m = r + hi8;
      hw[m * H_ + u + n] = (bf16_t)h;
      if (sid >= 0) {
        pool[r] += h;
        if (slast) {
          mean_vec[((size_t)m * N_ + sid) * H_ + u + n] = pool[r] / span_cnt[sid];
          pool[r] = 0.0f;
        }
      }
    }
  }
}

// ---------------- LayerNorm (optional residual), fp32 in -> fp32 (+bf16) out ----------------
__global__ __launch_bounds__(256) void ln_kernel(const float* __restrict__ in,
                                                 const float* __restrict__ resid,
                                                 const float* __restrict__ gamma,
                                                 const float* __restrict__ beta,
                                                 float* __restrict__ outF,
                                                 bf16_t* __restrict__ outB, int Hd) {
  __shared__ float red[256];
  __shared__ float stat[2];
  const int row = blockIdx.x;
  const float* x  = in + (size_t)row * Hd;
  const float* rz = resid ? resid + (size_t)row * Hd : nullptr;

  float s = 0.0f;
  for (int i = threadIdx.x; i < Hd; i += 256) s += x[i] + (rz ? rz[i] : 0.0f);
  red[threadIdx.x] = s; __syncthreads();
  for (int o = 128; o > 0; o >>= 1) { if (threadIdx.x < o) red[threadIdx.x] += red[threadIdx.x + o]; __syncthreads(); }
  if (threadIdx.x == 0) stat[0] = red[0] / Hd;
  __syncthreads();
  const float mean = stat[0];

  float s2 = 0.0f;
  for (int i = threadIdx.x; i < Hd; i += 256) {
    float v = x[i] + (rz ? rz[i] : 0.0f) - mean; s2 += v * v;
  }
  red[threadIdx.x] = s2; __syncthreads();
  for (int o = 128; o > 0; o >>= 1) { if (threadIdx.x < o) red[threadIdx.x] += red[threadIdx.x + o]; __syncthreads(); }
  if (threadIdx.x == 0) stat[1] = rsqrtf(red[0] / Hd + 1e-7f);
  __syncthreads();
  const float rstd = stat[1];

  for (int i = threadIdx.x; i < Hd; i += 256) {
    float v = (x[i] + (rz ? rz[i] : 0.0f) - mean) * rstd * gamma[i] + beta[i];
    size_t idx = (size_t)row * Hd + i;
    if (outF) outF[idx] = v;
    if (outB) outB[idx] = (bf16_t)v;
  }
}

// ---------------- tiny 32x32 attention per (batch, head) ----------------
__global__ __launch_bounds__(256) void attn_kernel(const float* __restrict__ q,
                                                   const float* __restrict__ k,
                                                   const float* __restrict__ v,
                                                   const int* __restrict__ amask,
                                                   bf16_t* __restrict__ ctx) {
  __shared__ float Qs[N_][DH_], Ks[N_][DH_], Vs[N_][DH_], Sc[N_][N_];
  const int b  = blockIdx.x >> 4;
  const int hh = blockIdx.x & 15;
  const size_t base = (size_t)b * N_ * H_ + hh * DH_;

  for (int i = threadIdx.x; i < N_ * DH_; i += 256) {
    int nn = i / DH_, d = i % DH_;
    size_t idx = base + (size_t)nn * H_ + d;
    Qs[nn][d] = q[idx]; Ks[nn][d] = k[idx]; Vs[nn][d] = v[idx];
  }
  __syncthreads();
  for (int p = threadIdx.x; p < N_ * N_; p += 256) {
    int qn = p / N_, kn = p % N_;
    float s = 0.0f;
    for (int d = 0; d < DH_; ++d) s += Qs[qn][d] * Ks[kn][d];
    s *= 0.125f;  // 1/sqrt(64)
    if (amask[b * N_ + qn] == 0 || amask[b * N_ + kn] == 0) s = -3.4028234e38f;
    Sc[qn][kn] = s;
  }
  __syncthreads();
  if (threadIdx.x < N_) {
    const int qn = threadIdx.x;
    float mx = Sc[qn][0];
    for (int j = 1; j < N_; ++j) mx = fmaxf(mx, Sc[qn][j]);
    float sum = 0.0f;
    for (int j = 0; j < N_; ++j) { float e = __expf(Sc[qn][j] - mx); Sc[qn][j] = e; sum += e; }
    const float inv = __builtin_amdgcn_rcpf(sum);
    for (int d = 0; d < DH_; ++d) {
      float a = 0.0f;
      for (int j = 0; j < N_; ++j) a += Sc[qn][j] * Vs[j][d];
      ctx[base + (size_t)qn * H_ + d] = (bf16_t)(a * inv);
    }
  }
}

// ---------------- classifier: out[row, l] = attn_out[row] . Wc[l] + bc[l], l < 3 ----------------
__global__ __launch_bounds__(256) void cls_kernel(const float* __restrict__ x,
                                                  const float* __restrict__ Wc,
                                                  const float* __restrict__ bc,
                                                  float* __restrict__ out) {
  __shared__ float red[256];
  const int row = blockIdx.x / 3;
  const int l   = blockIdx.x % 3;
  const float* xr = x  + (size_t)row * H_;
  const float* wr = Wc + (size_t)l * H_;
  float s = 0.0f;
  for (int i = threadIdx.x; i < H_; i += 256) s += xr[i] * wr[i];
  red[threadIdx.x] = s; __syncthreads();
  for (int o = 128; o > 0; o >>= 1) { if (threadIdx.x < o) red[threadIdx.x] += red[threadIdx.x + o]; __syncthreads(); }
  if (threadIdx.x == 0) out[row * 3 + l] = red[0] + bc[l];
}

// ---------------- host launcher ----------------
extern "C" void kernel_launch(void* const* d_in, const int* in_sizes, int n_in,
                              void* d_out, int out_size, void* d_ws, size_t ws_size,
                              hipStream_t stream) {
  (void)in_sizes; (void)n_in; (void)out_size; (void)ws_size;
  const float* x     = (const float*)d_in[0];
  const int*   head  = (const int*)d_in[1];
  const int*   tail  = (const int*)d_in[2];
  const int*   amask = (const int*)d_in[3];
  const float* W_ih  = (const float*)d_in[4];
  const float* W_hh  = (const float*)d_in[5];
  const float* b_ih  = (const float*)d_in[6];
  const float* b_hh  = (const float*)d_in[7];
  const float* ln_g  = (const float*)d_in[8];
  const float* ln_b  = (const float*)d_in[9];
  const float* Wq    = (const float*)d_in[10];
  const float* bq    = (const float*)d_in[11];
  const float* Wk    = (const float*)d_in[12];
  const float* bk    = (const float*)d_in[13];
  const float* Wv    = (const float*)d_in[14];
  const float* bv    = (const float*)d_in[15];
  const float* Wo    = (const float*)d_in[16];
  const float* bo    = (const float*)d_in[17];
  const float* ln2_g = (const float*)d_in[18];
  const float* ln2_b = (const float*)d_in[19];
  const float* Wc    = (const float*)d_in[20];
  const float* bc    = (const float*)d_in[21];

  char* ws = (char*)d_ws;
  unsigned* bar       = (unsigned*)(ws + WS_BAR);
  int*      span_id   = (int*)     (ws + WS_SID);
  int*      span_last = (int*)     (ws + WS_SLAST);
  float*    span_cnt  = (float*)   (ws + WS_SCNT);
  float*    bias_c    = (float*)   (ws + WS_BIAS);
  bf16_t*   hbuf      = (bf16_t*)  (ws + WS_H);
  float*    mean_vec  = (float*)   (ws + WS_MEAN);
  float*    xln       = (float*)   (ws + WS_XLN);
  bf16_t*   xlnb      = (bf16_t*)  (ws + WS_XLNB);
  float*    qb        = (float*)   (ws + WS_Q);
  float*    kb        = (float*)   (ws + WS_K);
  float*    vb        = (float*)   (ws + WS_V);
  bf16_t*   ctxb      = (bf16_t*)  (ws + WS_CTXB);
  float*    attn      = (float*)   (ws + WS_ATTN);
  float*    aout      = (float*)   (ws + WS_AOUT);
  bf16_t*   wqb       = (bf16_t*)  (ws + WS_WQB);
  bf16_t*   wkb       = (bf16_t*)  (ws + WS_WKB);
  bf16_t*   wvb       = (bf16_t*)  (ws + WS_WVB);
  bf16_t*   wob       = (bf16_t*)  (ws + WS_WOB);
  bf16_t*   wihb      = (bf16_t*)  (ws + WS_WIHB);
  bf16_t*   whhb      = (bf16_t*)  (ws + WS_WHHB);
  bf16_t*   xb        = (bf16_t*)  (ws + WS_XB);
  bf16_t*   xgb       = (bf16_t*)  (ws + WS_XGB);

  // init: barrier state, span tables, zero h double-buffer, zero mean_vec
  init_kernel<<<1, 1024, 0, stream>>>(head, tail, span_id, span_last, span_cnt, hbuf, bar);
  zero_f32_kernel<<<512, 256, 0, stream>>>(mean_vec, (size_t)B_ * N_ * H_);

  // fp32 -> bf16 conversions
  cvt_kernel<<<4096, 256, 0, stream>>>(x,    xb,   (size_t)B_ * S_ * H_);
  cvt_kernel<<<2048, 256, 0, stream>>>(W_ih, wihb, (size_t)G4_ * H_);
  cvt_kernel<<<2048, 256, 0, stream>>>(W_hh, whhb, (size_t)G4_ * H_);
  cvt_kernel<<<1024, 256, 0, stream>>>(Wq, wqb, (size_t)H_ * H_);
  cvt_kernel<<<1024, 256, 0, stream>>>(Wk, wkb, (size_t)H_ * H_);
  cvt_kernel<<<1024, 256, 0, stream>>>(Wv, wvb, (size_t)H_ * H_);
  cvt_kernel<<<1024, 256, 0, stream>>>(Wo, wob, (size_t)H_ * H_);
  bias_kernel<<<16, 256, 0, stream>>>(b_ih, b_hh, bias_c, G4_);

  // input projection: xg[B,S,4H] = x[B*S,H] * W_ih^T + (b_ih+b_hh)   (137 GFLOP, bf16 WMMA)
  wmma_gemm<1><<<dim3(1024, 32), 256, 0, stream>>>(xb, wihb, bias_c, nullptr, xgb,
                                                   B_ * S_, G4_, H_);

  // persistent LSTM recurrence (1024 steps, 1 grid barrier/step, pooled means on the fly)
  lstm_kernel<<<8, 256, 0, stream>>>(xgb, whhb, hbuf, mean_vec,
                                     span_id, span_last, span_cnt, bar, bar + 1);

  // LN1 over [512, 1024]
  ln_kernel<<<512, 256, 0, stream>>>(mean_vec, nullptr, ln_g, ln_b, xln, xlnb, H_);

  // QKV projections (bf16 WMMA, async-LDS A staging)
  wmma_gemm<0><<<dim3(32, 8), 256, 0, stream>>>(xlnb, wqb, bq, qb, nullptr, 512, H_, H_);
  wmma_gemm<0><<<dim3(32, 8), 256, 0, stream>>>(xlnb, wkb, bk, kb, nullptr, 512, H_, H_);
  wmma_gemm<0><<<dim3(32, 8), 256, 0, stream>>>(xlnb, wvb, bv, vb, nullptr, 512, H_, H_);

  // attention: 256 (batch, head) pairs of 32x32
  attn_kernel<<<256, 256, 0, stream>>>(qb, kb, vb, amask, ctxb);

  // output projection (bf16 WMMA), residual LN2, classifier
  wmma_gemm<0><<<dim3(32, 8), 256, 0, stream>>>(ctxb, wob, bo, attn, nullptr, 512, H_, H_);
  ln_kernel<<<512, 256, 0, stream>>>(attn, xln, ln2_g, ln2_b, aout, nullptr, H_);
  cls_kernel<<<1536, 256, 0, stream>>>(aout, Wc, bc, (float*)d_out);
}
